// LMHeadLossPipeLayer_48275432407494
// MI455X (gfx1250) — matrix operand order repaired and checked by reading, not compiled, not run on hardware
//
#include <hip/hip_runtime.h>
#include <hip/hip_bf16.h>
#include <math.h>

// ---------------------------------------------------------------------------
// Fused linear cross-entropy (LM head loss) for MI455X / gfx1250.
//   hidden: (1,2048,2048) fp32   labels: (1,2048) int64   weight: (151936,2048) fp32
//   out: scalar fp32 mean NLL over shifted tokens.
// Strategy: bf16 WMMA GEMM, weight streamed from HBM exactly once per element
// (panel-resident in 256KB LDS), online (max, sum-exp) per row, two-stage LSE merge.
// ---------------------------------------------------------------------------

typedef __attribute__((ext_vector_type(8)))  float  v8f;
typedef __attribute__((ext_vector_type(16))) __bf16 v16bf;

#define H_DIM       2048
#define NROWS       2048          // padded rows (row 2047 is zero padding)
#define VALID_ROWS  2047          // shifted sequence length
#define VOCAB       151936
#define NT          64            // vocab columns per workgroup panel
#define NPANELS     (VOCAB / NT)  // 2374 (exact)
#define MT          256           // rows per chunk = 8 waves * 32 rows
#define KT          32            // bf16 WMMA K
#define LDS_STRIDE  2056          // halves per LDS row (2048 + 8 pad -> bank spread)
#define IGNORE_IDX  (-100)

// ---- workspace layout (bytes) ----
#define OFF_HBF16 0                                     // 2048*2048*2  = 8388608
#define OFF_TGT   (8388608)                             // 2048*4       = 8192
#define OFF_PMAX  (8396800)                             // 2048*2374*4  = 19447808
#define OFF_PSUM  (8396800 + 2048*2374*4)
#define OFF_ACC   (8396800 + 2*2048*2374*4)             // 2 floats

__device__ __forceinline__ unsigned short f32_to_bf16_rne(float f) {
  unsigned int u = __float_as_uint(f);
  u += 0x7FFFu + ((u >> 16) & 1u);                      // round-to-nearest-even
  return (unsigned short)(u >> 16);
}

// ---------------------------------------------------------------------------
// Phase 1: hidden fp32 -> bf16 (rows 0..2046 = shift_hidden; row 2047 zeroed).
// Also zero-init the global accumulators.
// ---------------------------------------------------------------------------
__global__ void cvt_hidden_kernel(const float* __restrict__ hidden,
                                  unsigned short* __restrict__ hbf,
                                  float* __restrict__ acc) {
  const size_t idx = (size_t)blockIdx.x * blockDim.x + threadIdx.x;  // one float4
  if (idx == 0) { acc[0] = 0.0f; acc[1] = 0.0f; }
  const size_t row = idx >> 9;                                       // 512 float4 per row
  const size_t c4  = idx & 511;
  if (row >= NROWS) return;
  unsigned int lo = 0u, hi = 0u;
  if (row < VALID_ROWS) {
    const float4 v = ((const float4*)(hidden + row * H_DIM))[c4];
    lo = (unsigned)f32_to_bf16_rne(v.x) | ((unsigned)f32_to_bf16_rne(v.y) << 16);
    hi = (unsigned)f32_to_bf16_rne(v.z) | ((unsigned)f32_to_bf16_rne(v.w) << 16);
  }
  uint2 o; o.x = lo; o.y = hi;
  ((uint2*)(hbf + row * H_DIM))[c4] = o;
}

// ---------------------------------------------------------------------------
// Phase 2: one workgroup per 64-column vocab panel.
//   - convert 64x2048 fp32 weight panel to bf16 in LDS (read once from HBM)
//   - for each 256-row chunk: bf16 WMMA GEMM, then per-row (max, sum-exp)
//   - write per-panel partials + target logits
// ---------------------------------------------------------------------------
__global__ void __launch_bounds__(256)
fused_ce_panel_kernel(const unsigned short* __restrict__ hbf,
                      const float* __restrict__ weight,
                      const long long* __restrict__ labels,
                      float* __restrict__ pmax,
                      float* __restrict__ psum,
                      float* __restrict__ tgt) {
  extern __shared__ __align__(16) unsigned short wtile[];   // [NT][LDS_STRIDE] bf16
  const int tid   = threadIdx.x;
  const int panel = blockIdx.x;
  const int v0    = panel * NT;

  // ---- load + convert weight panel: 64 rows x 2048 K, fp32 -> bf16 LDS ----
  for (int idx = tid; idx < NT * (H_DIM / 4); idx += 256) {
    const int r  = idx >> 9;                // 512 float4 per row
    const int c4 = idx & 511;
    const float4 wv = ((const float4*)(weight + (size_t)(v0 + r) * H_DIM))[c4];
    unsigned int lo = (unsigned)f32_to_bf16_rne(wv.x) | ((unsigned)f32_to_bf16_rne(wv.y) << 16);
    unsigned int hi = (unsigned)f32_to_bf16_rne(wv.z) | ((unsigned)f32_to_bf16_rne(wv.w) << 16);
    uint2 o; o.x = lo; o.y = hi;
    *((uint2*)(wtile + r * LDS_STRIDE + c4 * 4)) = o;
  }
  __syncthreads();

  const int lane    = tid & 31;
  const int wave    = tid >> 5;
  const int m15     = lane & 15;
  const int halfsel = lane >> 4;            // 0: lanes 0-15, 1: lanes 16-31

  for (int chunk = 0; chunk < NROWS / MT; ++chunk) {
    const int rowbase = chunk * MT + wave * 32;   // this wave: rows [rowbase, rowbase+32)

    v8f c[2][4];
    const v8f zero8 = {0.f, 0.f, 0.f, 0.f, 0.f, 0.f, 0.f, 0.f};
    #pragma unroll
    for (int g = 0; g < 2; ++g)
      #pragma unroll
      for (int t = 0; t < 4; ++t) c[g][t] = zero8;

    // ---- K loop: A from global bf16 hidden, B from LDS panel ----
    for (int k0 = 0; k0 < H_DIM; k0 += KT) {
      v16bf a[2];
      #pragma unroll
      for (int g = 0; g < 2; ++g) {
        // 16-bit A 16x32 layout: lane holds row (lane&15); K chunks
        // [k0 + 8*half, +8) in v0-3 and [k0 + 16 + 8*half, +8) in v4-7.
        const unsigned short* ap =
            hbf + (size_t)(rowbase + g * 16 + m15) * H_DIM + k0 + halfsel * 8;
        union { uint4 q[2]; v16bf v; } ua;
        ua.q[0] = *((const uint4*)ap);
        ua.q[1] = *((const uint4*)(ap + 16));
        a[g] = ua.v;
      }
      #pragma unroll
      for (int t = 0; t < 4; ++t) {
        // 16-bit B 32x16 layout: lane holds column (lane&15); K chunks
        // [k0 + 8*half, +8) in v0-3 and [k0 + 16 + 8*half, +8) in v4-7.
        const unsigned short* bp =
            wtile + (t * 16 + m15) * LDS_STRIDE + k0 + halfsel * 8;
        union { uint4 q[2]; v16bf v; } ub;
        ub.q[0] = *((const uint4*)bp);
        ub.q[1] = *((const uint4*)(bp + 16));
        const v16bf b = ub.v;
        #pragma unroll
        for (int g = 0; g < 2; ++g)
          c[g][t] = __builtin_amdgcn_wmma_f32_16x16x32_bf16(
              false, a[g], false, b, (short)0, c[g][t], false, false);
      }
    }

    // ---- epilogue: per-row max / sum-exp over the 64 panel columns ----
    // C layout: VGPR r -> row (rowbase + g*16 + 8*half + r), column (t*16 + lane&15).
    #pragma unroll
    for (int g = 0; g < 2; ++g) {
      #pragma unroll
      for (int r = 0; r < 8; ++r) {
        float m = c[g][0][r];
        #pragma unroll
        for (int t = 1; t < 4; ++t) m = fmaxf(m, c[g][t][r]);
        #pragma unroll
        for (int off = 1; off < 16; off <<= 1) m = fmaxf(m, __shfl_xor(m, off, 32));
        float s = 0.0f;
        #pragma unroll
        for (int t = 0; t < 4; ++t) s += __expf(c[g][t][r] - m);
        #pragma unroll
        for (int off = 1; off < 16; off <<= 1) s += __shfl_xor(s, off, 32);

        const int row = rowbase + g * 16 + halfsel * 8 + r;
        if (m15 == 0) {
          pmax[(size_t)row * NPANELS + panel] = m;
          psum[(size_t)row * NPANELS + panel] = s;
        }
        // target logit extraction (shifted labels: label for row i is labels[i+1])
        if (row < VALID_ROWS) {
          const int lab = (int)labels[row + 1];
          const int lc  = lab - v0;
          if (lc >= 0 && lc < NT && (lc & 15) == m15) {
            #pragma unroll
            for (int t = 0; t < 4; ++t)
              if ((lc >> 4) == t) tgt[row] = c[g][t][r];
          }
        }
      }
    }
  }
}

// ---------------------------------------------------------------------------
// Phase 3: one wave per row -> merge 2374 partials into logsumexp, accumulate
// masked NLL sum and valid count.
// ---------------------------------------------------------------------------
__global__ void lse_nll_kernel(const float* __restrict__ pmax,
                               const float* __restrict__ psum,
                               const float* __restrict__ tgt,
                               const long long* __restrict__ labels,
                               float* __restrict__ acc) {
  const int row  = blockIdx.x * 8 + (threadIdx.x >> 5);
  const int lane = threadIdx.x & 31;
  if (row >= VALID_ROWS) return;
  const float* pm = pmax + (size_t)row * NPANELS;
  const float* ps = psum + (size_t)row * NPANELS;

  float m = -3.402823466e+38f;
  for (int p = lane; p < NPANELS; p += 32) m = fmaxf(m, pm[p]);
  #pragma unroll
  for (int off = 1; off < 32; off <<= 1) m = fmaxf(m, __shfl_xor(m, off, 32));

  float s = 0.0f;
  for (int p = lane; p < NPANELS; p += 32) s += ps[p] * __expf(pm[p] - m);
  #pragma unroll
  for (int off = 1; off < 32; off <<= 1) s += __shfl_xor(s, off, 32);

  if (lane == 0) {
    const int  lab   = (int)labels[row + 1];
    const bool valid = (lab != IGNORE_IDX);
    const float nll  = valid ? (m + logf(s) - tgt[row]) : 0.0f;
    atomicAdd(&acc[0], nll);
    if (valid) atomicAdd(&acc[1], 1.0f);
  }
}

// Phase 4: scalar mean.
__global__ void finalize_kernel(const float* __restrict__ acc, float* __restrict__ out) {
  out[0] = acc[0] / acc[1];
}

// ---------------------------------------------------------------------------
extern "C" void kernel_launch(void* const* d_in, const int* in_sizes, int n_in,
                              void* d_out, int out_size, void* d_ws, size_t ws_size,
                              hipStream_t stream) {
  (void)in_sizes; (void)n_in; (void)out_size; (void)ws_size;
  const float*     hidden = (const float*)d_in[0];
  const long long* labels = (const long long*)d_in[1];
  const float*     weight = (const float*)d_in[2];

  char* ws = (char*)d_ws;
  unsigned short* hbf  = (unsigned short*)(ws + OFF_HBF16);
  float*          tgt  = (float*)(ws + OFF_TGT);
  float*          pmax = (float*)(ws + OFF_PMAX);
  float*          psum = (float*)(ws + OFF_PSUM);
  float*          acc  = (float*)(ws + OFF_ACC);

  // Phase 1: hidden fp32 -> bf16 (+shift pad, zero accumulators)
  cvt_hidden_kernel<<<(NROWS * (H_DIM / 4) + 255) / 256, 256, 0, stream>>>(hidden, hbf, acc);

  // Phase 2: panel GEMM + online softmax stats (256KB dynamic LDS per WG)
  const size_t lds_bytes = (size_t)NT * LDS_STRIDE * sizeof(unsigned short); // 263168
  (void)hipFuncSetAttribute((const void*)fused_ce_panel_kernel,
                            hipFuncAttributeMaxDynamicSharedMemorySize, (int)lds_bytes);
  fused_ce_panel_kernel<<<NPANELS, 256, lds_bytes, stream>>>(hbf, weight, labels,
                                                             pmax, psum, tgt);

  // Phase 3: per-row logsumexp merge + NLL accumulation
  lse_nll_kernel<<<(VALID_ROWS + 7) / 8, 256, 0, stream>>>(pmax, psum, tgt, labels, acc);

  // Phase 4: mean
  finalize_kernel<<<1, 1, 0, stream>>>(acc, (float*)d_out);
}
